// MultiHeadSelfAttention_14748917694848
// MI455X (gfx1250) — matrix operand order repaired and checked by reading, not compiled, not run on hardware
//
#include <hip/hip_runtime.h>
#include <hip/hip_bf16.h>

typedef __attribute__((ext_vector_type(16))) __bf16 v16bf;
typedef __attribute__((ext_vector_type(8)))  float  v8f;

#define BN   4
#define SEQ  4096
#define DIM  512
#define NH   8
#define HDIM 64
#define ROWS (BN * SEQ)   // 16384

static __device__ __forceinline__ v8f wmma_bf16(v16bf a, v16bf b, v8f c) {
  // D = A(16x32 bf16) * B(32x16 bf16) + C(16x16 f32)
  return __builtin_amdgcn_wmma_f32_16x16x32_bf16(false, a, false, b, (short)0, c,
                                                 false, false);
}

// ---------------------------------------------------------------------------
// Kernel 1: fused Q/K/V projection GEMM.  out[bh] = x @ W^T + bias (bf16 out)
// grid = (ROWS/64, DIM/32, 3), block = 256 (8 waves, 4x2 wave tile grid)
// ---------------------------------------------------------------------------
__global__ __launch_bounds__(256) void qkv_proj_kernel(
    const float* __restrict__ x,
    const float* __restrict__ Wq, const float* __restrict__ bq,
    const float* __restrict__ Wk, const float* __restrict__ bk,
    const float* __restrict__ Wv, const float* __restrict__ bv,
    __bf16* __restrict__ qout, __bf16* __restrict__ kout,
    __bf16* __restrict__ vout)
{
  const float* W; const float* bias; __bf16* out;
  if (blockIdx.z == 0)      { W = Wq; bias = bq; out = qout; }
  else if (blockIdx.z == 1) { W = Wk; bias = bk; out = kout; }
  else                      { W = Wv; bias = bv; out = vout; }

  __shared__ __bf16 sX[64][40];   // 64 rows x 32 k (padded)
  __shared__ __bf16 sW[32][40];   // 32 out-features x 32 k (padded)

  const int t     = threadIdx.x;
  const int lane  = t & 31;
  const int wave  = t >> 5;
  const int waveM = wave >> 1;          // 0..3
  const int waveN = wave & 1;           // 0..1
  const int m0    = blockIdx.x * 64;
  const int n0    = blockIdx.y * 32;
  const int hi8   = (lane >= 16) ? 8 : 0;
  const int hi16  = (lane >= 16) ? 16 : 0;

  v8f c = {};

  for (int kt = 0; kt < DIM; kt += 32) {
    // stage X tile (fp32 -> bf16), coalesced 32 cols per row-group
    #pragma unroll
    for (int i = 0; i < 8; ++i) {
      int idx = i * 256 + t;
      int r = idx >> 5, cc = idx & 31;
      sX[r][cc] = (__bf16)x[(size_t)(m0 + r) * DIM + kt + cc];
    }
    // stage W tile (row = output feature)
    #pragma unroll
    for (int i = 0; i < 4; ++i) {
      int idx = i * 256 + t;
      int r = idx >> 5, cc = idx & 31;
      sW[r][cc] = (__bf16)W[(size_t)(n0 + r) * DIM + kt + cc];
    }
    __syncthreads();

    // A fragment: 16x32 bf16, m = lane&15 striped over lanes
    v16bf a, b;
    const int am = waveM * 16 + (lane & 15);
    #pragma unroll
    for (int v = 0; v < 8; ++v)
      #pragma unroll
      for (int p = 0; p < 2; ++p) {
        int kk = 2 * v + p + hi16;
        a[v * 2 + p] = sX[am][kk];
      }
    // B fragment: 32x16 bf16, B[k][n] = W[n][k]
    const int bnr = waveN * 16 + (lane & 15);
    #pragma unroll
    for (int i = 0; i < 16; ++i)
      b[i] = sW[bnr][i + hi16];

    c = wmma_bf16(a, b, c);
    __syncthreads();
  }

  // epilogue: bias + bf16 store, head-major layout [B][H][N][HD]
  const int col = n0 + waveN * 16 + (lane & 15);
  const float bval = bias[col];
  const int h = col >> 6, hd = col & 63;
  #pragma unroll
  for (int r = 0; r < 8; ++r) {
    int m  = m0 + waveM * 16 + r + hi8;   // row in [0, ROWS)
    int bb = m >> 12;                      // batch
    int n  = m & 4095;                     // seq pos
    float val = c[r] + bval;
    out[(((size_t)(bb * NH + h) * SEQ + n) * HDIM) + hd] = (__bf16)val;
  }
}

// ---------------------------------------------------------------------------
// Kernel 2: flash attention over one (b,h); each wave owns 16 query rows.
// grid = (SEQ/128, B*H), block = 256 (8 waves)
// ---------------------------------------------------------------------------
__global__ __launch_bounds__(256) void flash_attn_kernel(
    const __bf16* __restrict__ qws, const __bf16* __restrict__ kws,
    const __bf16* __restrict__ vws, __bf16* __restrict__ ows)
{
  __shared__ __bf16 sK[32][72];        // 32 K-rows x 64 hd (padded)
  __shared__ __bf16 sV[32][72];        // 32 V-rows x 64 hd (padded)
  __shared__ __bf16 sP[8][16][40];     // per-wave P tile (C->A relayout)

  const int t    = threadIdx.x;
  const int lane = t & 31;
  const int wave = t >> 5;
  const int hi8  = (lane >= 16) ? 8 : 0;
  const int hi16 = (lane >= 16) ? 16 : 0;
  const int bh   = blockIdx.y;              // b*8 + h
  const int b    = bh >> 3, h = bh & 7;
  const int q0   = blockIdx.x * 128 + wave * 16;

  const __bf16* qp = qws + (size_t)bh * SEQ * HDIM;
  const __bf16* kp = kws + (size_t)bh * SEQ * HDIM;
  const __bf16* vp = vws + (size_t)bh * SEQ * HDIM;

  // Q resident as two A-fragments (hd 0..31 and 32..63)
  v16bf aq0, aq1;
  {
    const int m = lane & 15;
    #pragma unroll
    for (int v = 0; v < 8; ++v)
      #pragma unroll
      for (int p = 0; p < 2; ++p) {
        int kk = 2 * v + p + hi16;
        aq0[v * 2 + p] = qp[(size_t)(q0 + m) * HDIM + kk];
        aq1[v * 2 + p] = qp[(size_t)(q0 + m) * HDIM + 32 + kk];
      }
  }

  float mrun[8], lrun[8];
  #pragma unroll
  for (int r = 0; r < 8; ++r) { mrun[r] = -3.0e38f; lrun[r] = 0.0f; }
  v8f acc[4] = {{}, {}, {}, {}};

  const float sc = 0.125f;   // 1/sqrt(64)

  for (int k0 = 0; k0 < SEQ; k0 += 32) {
    // stage K/V tiles (bf16, already converted), coalesced
    #pragma unroll
    for (int i = 0; i < 8; ++i) {
      int idx = i * 256 + t;
      int r = idx >> 6, cc = idx & 63;
      sK[r][cc] = kp[(size_t)(k0 + r) * HDIM + cc];
      sV[r][cc] = vp[(size_t)(k0 + r) * HDIM + cc];
    }
    if (k0 + 32 < SEQ) {
      __builtin_prefetch(kp + (size_t)(k0 + 32) * HDIM + lane * 8, 0, 0);
      __builtin_prefetch(vp + (size_t)(k0 + 32) * HDIM + lane * 8, 0, 0);
    }
    __syncthreads();

    // S = Q @ K^T : two 16x16 score tiles (K rows k0..k0+15 and +16..+31)
    v8f s0 = {}, s1 = {};
    #pragma unroll
    for (int sub = 0; sub < 2; ++sub) {
      v16bf b0, b1;
      const int kr = sub * 16 + (lane & 15);   // B column n = K-row
      #pragma unroll
      for (int i = 0; i < 16; ++i) {
        b0[i] = sK[kr][i + hi16];              // hd 0..31
        b1[i] = sK[kr][32 + i + hi16];         // hd 32..63
      }
      if (sub == 0) { s0 = wmma_bf16(aq0, b0, s0); s0 = wmma_bf16(aq1, b1, s0); }
      else          { s1 = wmma_bf16(aq0, b0, s1); s1 = wmma_bf16(aq1, b1, s1); }
    }

    // online softmax (row reductions across 16-lane halves)
    #pragma unroll
    for (int r = 0; r < 8; ++r) {
      float v0 = s0[r] * sc, v1 = s1[r] * sc;
      float rmax = fmaxf(v0, v1);
      #pragma unroll
      for (int off = 1; off < 16; off <<= 1)
        rmax = fmaxf(rmax, __shfl_xor(rmax, off, 16));
      float mnew = fmaxf(mrun[r], rmax);
      float f  = __expf(mrun[r] - mnew);
      float p0 = __expf(v0 - mnew);
      float p1 = __expf(v1 - mnew);
      float rs = p0 + p1;
      #pragma unroll
      for (int off = 1; off < 16; off <<= 1)
        rs += __shfl_xor(rs, off, 16);
      lrun[r] = lrun[r] * f + rs;
      mrun[r] = mnew;
      acc[0][r] *= f; acc[1][r] *= f; acc[2][r] *= f; acc[3][r] *= f;
      // C-layout -> LDS (wave-private), row m = r + hi8, cols = score col
      sP[wave][r + hi8][lane & 15]        = (__bf16)p0;
      sP[wave][r + hi8][16 + (lane & 15)] = (__bf16)p1;
    }
    asm volatile("s_wait_dscnt 0" ::: "memory");

    // re-read P in A-fragment layout (16x32)
    v16bf pa;
    {
      const int m = lane & 15;
      #pragma unroll
      for (int v = 0; v < 8; ++v)
        #pragma unroll
        for (int p = 0; p < 2; ++p) {
          int kk = 2 * v + p + hi16;
          pa[v * 2 + p] = sP[wave][m][kk];
        }
    }

    // acc += P @ V   (4 WMMAs cover hd 0..63)
    #pragma unroll
    for (int j = 0; j < 4; ++j) {
      v16bf bv;
      const int n = j * 16 + (lane & 15);
      #pragma unroll
      for (int i = 0; i < 16; ++i)
        bv[i] = sV[i + hi16][n];               // B[k][n] = V[krow][hd]
      acc[j] = wmma_bf16(pa, bv, acc[j]);
    }
    __syncthreads();
  }

  // normalize + store, layout [B][N][D] with d = h*64 + hd (bf16)
  #pragma unroll
  for (int r = 0; r < 8; ++r) {
    float inv = 1.0f / lrun[r];
    int m = q0 + r + hi8;
    size_t base = ((size_t)(b * SEQ + m)) * DIM + h * HDIM;
    #pragma unroll
    for (int j = 0; j < 4; ++j)
      ows[base + j * 16 + (lane & 15)] = (__bf16)(acc[j][r] * inv);
  }
}

// ---------------------------------------------------------------------------
// Kernel 3: output projection, fp32 out.  out = attn @ Wo^T + bo
// grid = (ROWS/64, DIM/32), block = 256
// ---------------------------------------------------------------------------
__global__ __launch_bounds__(256) void out_proj_kernel(
    const __bf16* __restrict__ ain, const float* __restrict__ Wo,
    const float* __restrict__ bo, float* __restrict__ out)
{
  __shared__ __bf16 sX[64][40];
  __shared__ __bf16 sW[32][40];

  const int t     = threadIdx.x;
  const int lane  = t & 31;
  const int wave  = t >> 5;
  const int waveM = wave >> 1;
  const int waveN = wave & 1;
  const int m0    = blockIdx.x * 64;
  const int n0    = blockIdx.y * 32;
  const int hi8   = (lane >= 16) ? 8 : 0;
  const int hi16  = (lane >= 16) ? 16 : 0;

  v8f c = {};

  for (int kt = 0; kt < DIM; kt += 32) {
    #pragma unroll
    for (int i = 0; i < 8; ++i) {
      int idx = i * 256 + t;
      int r = idx >> 5, cc = idx & 31;
      sX[r][cc] = ain[(size_t)(m0 + r) * DIM + kt + cc];
    }
    #pragma unroll
    for (int i = 0; i < 4; ++i) {
      int idx = i * 256 + t;
      int r = idx >> 5, cc = idx & 31;
      sW[r][cc] = (__bf16)Wo[(size_t)(n0 + r) * DIM + kt + cc];
    }
    __syncthreads();

    v16bf a, b;
    const int am = waveM * 16 + (lane & 15);
    #pragma unroll
    for (int v = 0; v < 8; ++v)
      #pragma unroll
      for (int p = 0; p < 2; ++p) {
        int kk = 2 * v + p + hi16;
        a[v * 2 + p] = sX[am][kk];
      }
    const int bnr = waveN * 16 + (lane & 15);
    #pragma unroll
    for (int i = 0; i < 16; ++i)
      b[i] = sW[bnr][i + hi16];

    c = wmma_bf16(a, b, c);
    __syncthreads();
  }

  const int col = n0 + waveN * 16 + (lane & 15);
  const float bval = bo[col];
  #pragma unroll
  for (int r = 0; r < 8; ++r) {
    int m = m0 + waveM * 16 + r + hi8;
    out[(size_t)m * DIM + col] = c[r] + bval;
  }
}

// ---------------------------------------------------------------------------
extern "C" void kernel_launch(void* const* d_in, const int* in_sizes, int n_in,
                              void* d_out, int out_size, void* d_ws,
                              size_t ws_size, hipStream_t stream)
{
  const float* x  = (const float*)d_in[0];
  const float* Wq = (const float*)d_in[1];
  const float* bq = (const float*)d_in[2];
  const float* Wk = (const float*)d_in[3];
  const float* bk = (const float*)d_in[4];
  const float* Wv = (const float*)d_in[5];
  const float* bv = (const float*)d_in[6];
  const float* Wo = (const float*)d_in[7];
  const float* bo = (const float*)d_in[8];

  const size_t elems = (size_t)ROWS * DIM;   // 8,388,608
  __bf16* qws = (__bf16*)d_ws;
  __bf16* kws = qws + elems;
  __bf16* vws = kws + elems;
  __bf16* aws = vws + elems;                 // total 64 MB of workspace

  dim3 g1(ROWS / 64, DIM / 32, 3);
  qkv_proj_kernel<<<g1, 256, 0, stream>>>(x, Wq, bq, Wk, bk, Wv, bv,
                                          qws, kws, vws);

  dim3 g2(SEQ / 128, BN * NH);
  flash_attn_kernel<<<g2, 256, 0, stream>>>(qws, kws, vws, aws);

  dim3 g3(ROWS / 64, DIM / 32);
  out_proj_kernel<<<g3, 256, 0, stream>>>(aws, Wo, bo, (float*)d_out);
}
